// EdgeAwareAttnLayer_55190329753792
// MI455X (gfx1250) — compile-verified, matrix operand order checked by reading, not run on hardware
//
#include <hip/hip_runtime.h>
#include <hip/hip_bf16.h>
#include <math.h>

// ---------------------------------------------------------------------------
// EdgeAwareAttnLayer for MI455X (gfx1250, wave32, WMMA).
// Memory-bound (~0.6-1 GB traffic -> ~30-45us at 23.3 TB/s). All GEMMs
// (QKV proj, edge MLP 38->64->32->8, output proj) run on
// v_wmma_f32_16x16x32_bf16. Weights are pre-packed once per launch into
// fragment-ordered bf16 so B loads are single unconditional 32B vector loads;
// A tiles are staged in LDS pre-swizzled to the A-fragment layout so A loads
// are plain ds_load_b128 pairs. Full-tile fast paths keep stores/loads free of
// per-lane exec masking. Segment reductions use global f32 atomics.
// Softmax max-subtraction is dropped: e = tanh(.) in [-1,1] makes
// exp(e)/sum(exp(e)) exactly the reference softmax.
// ---------------------------------------------------------------------------

typedef __attribute__((ext_vector_type(16))) __bf16 v16bf;
typedef __attribute__((ext_vector_type(8)))  float  v8f;

#define HB    8     // heads
#define DKH   16    // dims per head
#define OUTD  128
#define IND   128
#define LREW  32    // lr embedding width
#define FIJV  38    // valid Fij features
#define FIJW  64    // Fij padded to WMMA K granularity
#define EMHW  64    // hidden width of edge MLP
#define EBDW  32    // edge embedding width

__device__ __forceinline__ __bf16 f32_bf16(float f) {
  union { float f; unsigned u; } x; x.f = f;
  unsigned r = x.u + 0x7FFFu + ((x.u >> 16) & 1u);   // round-to-nearest-even
  unsigned short h = (unsigned short)(r >> 16);
  return __builtin_bit_cast(__bf16, h);
}

// 16-bit A/B fragment K offset for element i, lane-half `half` (ISA 7.12.2):
// i in [0,8):  k = half*8 + i ; i in [8,16): k = 16 + half*8 + (i-8)
__device__ __forceinline__ int kmap(int i, int half) {
  return ((i < 8) ? 0 : 16) + half * 8 + (i & 7);
}

// Inverse map: element index within a 32-wide K group for K offset kr.
__device__ __forceinline__ void kinv(int kr, int& half, int& i) {
  if (kr < 16) { half = kr >> 3; i = kr & 7; }
  else         { half = (kr - 16) >> 3; i = 8 + ((kr - 16) & 7); }
}

// Swizzled A-tile index for element (m, k): tiles stored [kt][lane][16] bf16.
__device__ __forceinline__ int a_swz(int m, int k) {
  int half, i; kinv(k & 31, half, i);
  return ((k >> 5) * 32 + (half << 4) + m) * 16 + i;
}

__device__ __forceinline__ v8f wmma_bf16(v16bf a, v16bf b, v8f c) {
  return __builtin_amdgcn_wmma_f32_16x16x32_bf16(
      false, a, false, b, (short)0, c, false, false);
}

// ---------------------------------------------------------------------------
// K0: pack W (row-major nmax x ldw, used as B = W^T) into fragment order:
//     out[((nt*ktiles + kt)*32 + lane)*16 + i] = bf16(W[n][k]) zero-padded.
// ---------------------------------------------------------------------------
__global__ void pack_b_kernel(const float* __restrict__ W, int ldw, int kmax, int nmax,
                              int ktiles, int ntiles, __bf16* __restrict__ out) {
  int idx = blockIdx.x * blockDim.x + threadIdx.x;
  int total = ntiles * ktiles * 512;
  if (idx >= total) return;
  int i    = idx & 15;
  int lane = (idx >> 4) & 31;
  int kt   = (idx >> 9) % ktiles;
  int nt   = idx / (ktiles * 512);
  int n = nt * 16 + (lane & 15);
  int k = kt * 32 + kmap(i, lane >> 4);
  float f = (k < kmax && n < nmax) ? W[n * ldw + k] : 0.0f;
  out[idx] = f32_bf16(f);
}

// ---------------------------------------------------------------------------
// Shared helper for node-side GEMMs: stage a 16 x 128 f32 tile into bf16
// A-fragment order in LDS, then 4 WMMAs against packed B per wave.
// ---------------------------------------------------------------------------
__device__ __forceinline__ void stage_a_tile(const float* __restrict__ src, int tile,
                                             int n_nodes, float* xs, __bf16* as, int tid) {
  if (tile * 16 + 16 <= n_nodes) {                              // full tile: no guards
    for (int i = tid; i < 16 * IND; i += 256)
      xs[i] = src[(size_t)tile * 16 * IND + i];
  } else {
    for (int i = tid; i < 16 * IND; i += 256) {
      int row = tile * 16 + (i >> 7);
      xs[i] = (row < n_nodes) ? src[(size_t)tile * 16 * IND + i] : 0.f;
    }
  }
  __syncthreads();
  for (int idx = tid; idx < 16 * IND; idx += 256) {             // swizzle -> bf16
    int i = idx & 15, lane = (idx >> 4) & 31, kt = idx >> 9;
    int m = lane & 15, kk = kt * 32 + kmap(i, lane >> 4);
    as[idx] = f32_bf16(xs[m * IND + kk]);
  }
  __syncthreads();
}

// ---------------------------------------------------------------------------
// K1: q/k/v = x @ W{q,k,v}^T.   grid = (N/16, 3), block = 256 (8 waves).
// ---------------------------------------------------------------------------
__global__ void __launch_bounds__(256)
qkv_gemm_kernel(const float* __restrict__ x,
                const __bf16* __restrict__ Bq, const __bf16* __restrict__ Bk,
                const __bf16* __restrict__ Bv,
                float* __restrict__ q, float* __restrict__ k, float* __restrict__ v,
                int n_nodes) {
  __shared__ float xs[16 * IND];
  __shared__ __attribute__((aligned(32))) __bf16 as[16 * IND];  // [kt][lane][16]
  int tile = blockIdx.x, tid = threadIdx.x;
  const __bf16* B = (blockIdx.y == 0) ? Bq : (blockIdx.y == 1 ? Bk : Bv);
  float*       ot = (blockIdx.y == 0) ? q  : (blockIdx.y == 1 ? k  : v);

  // Prefetch next tile of x while this one is in flight (global_prefetch_b8).
  if (tile + 1 < gridDim.x)
    __builtin_prefetch(x + (size_t)(tile + 1) * 16 * IND + tid * 8, 0, 1);

  stage_a_tile(x, tile, n_nodes, xs, as, tid);

  int wave = tid >> 5, lane = tid & 31;
  v8f acc = {};
#pragma unroll
  for (int kt = 0; kt < IND / 32; ++kt) {
    v16bf a = *(const v16bf*)(as + (kt * 32 + lane) * 16);
    v16bf b = *(const v16bf*)(B + (((size_t)wave * (IND / 32) + kt) * 32 + lane) * 16);
    acc = wmma_bf16(a, b, acc);
  }
  int m0 = (lane >> 4) * 8, n = wave * 16 + (lane & 15);
  if (tile * 16 + 16 <= n_nodes) {                 // uniform full-tile fast path
    float* orow = ot + (size_t)(tile * 16 + m0) * OUTD + n;
#pragma unroll
    for (int r = 0; r < 8; ++r) orow[(size_t)r * OUTD] = acc[r];
  } else {
#pragma unroll
    for (int r = 0; r < 8; ++r) {
      int row = tile * 16 + m0 + r;
      if (row < n_nodes) ot[(size_t)row * OUTD + n] = acc[r];
    }
  }
}

// ---------------------------------------------------------------------------
// K2: per-edge MLP (WMMA chain through wave-private A-swizzled LDS tiles) +
//     q.k dots + e_raw and dst-segment moment atomics. 16 edges per wave.
// ---------------------------------------------------------------------------
__global__ void __launch_bounds__(256)
edge_mlp_dot_kernel(const int* __restrict__ edge_index,
                    const float* __restrict__ ea,
                    const float* __restrict__ lr_emb,
                    const __bf16* __restrict__ B1, const float* __restrict__ b1,
                    const __bf16* __restrict__ B2, const float* __restrict__ b2,
                    const __bf16* __restrict__ Bwb,
                    const float* __restrict__ q, const float* __restrict__ kmat,
                    const float* __restrict__ p_sigma, const int* __restrict__ p_log1p,
                    const float* __restrict__ p_cscale, const float* __restrict__ p_tau,
                    float* __restrict__ e_raw,
                    float* __restrict__ s1, float* __restrict__ s2, float* __restrict__ cnt,
                    int n_edges) {
  __shared__ __attribute__((aligned(32))) __bf16 s_fij[8][16 * FIJW];  // 16 KB
  __shared__ __attribute__((aligned(32))) __bf16 s_h1 [8][16 * EMHW];  // 16 KB
  __shared__ __attribute__((aligned(32))) __bf16 s_phi[8][16 * EBDW];  //  8 KB
  int tid = threadIdx.x, wave = tid >> 5, lane = tid & 31;
  int e0 = (blockIdx.x * 8 + wave) * 16;
  bool full = (e0 + 16 <= n_edges);                // wave-uniform
  __bf16* fij = s_fij[wave];
  __bf16* h1  = s_h1[wave];
  __bf16* phi = s_phi[wave];

  float sigma  = fmaxf(p_sigma[0], 1e-6f);
  float cscale = fmaxf(p_cscale[0], 1e-6f);
  int   lg     = p_log1p[0];
  float tau    = fmaxf(p_tau[0], 1e-6f);

  // ---- zero Fij tile (covers K padding), then fill valid features ----
  {
    uint4* z = (uint4*)fij;                        // 2048 B = 128 uint4
#pragma unroll
    for (int t = 0; t < 4; ++t) z[lane + 32 * t] = make_uint4(0, 0, 0, 0);
  }
  if (lane < 16) {
    int e = e0 + lane;
    bool ok = e < n_edges;
    float dist = ok ? ea[(size_t)e * 7 + 0] : 0.f;
    float co   = ok ? ea[(size_t)e * 7 + 1] : 0.f;
    float f0 = __expf(-dist / sigma);
    float f1 = lg ? (log1pf(fmaxf(co, 0.f)) / cscale) : (co / cscale);
    fij[a_swz(lane, 0)] = f32_bf16(ok ? f0 : 0.f);
    fij[a_swz(lane, 1)] = f32_bf16(ok ? f1 : 0.f);
#pragma unroll
    for (int j = 0; j < 4; ++j) {
      float c = ok ? fminf(fmaxf(ea[(size_t)e * 7 + 3 + j], 0.f), 1.f) : 0.f;
      fij[a_swz(lane, 2 + j)] = f32_bf16(c);
    }
  }
  for (int idx = lane; idx < 16 * LREW; idx += 32) {
    int el = idx >> 5, j = idx & 31;
    int e = e0 + el;
    float val = 0.f;
    if (e < n_edges) {
      int lrid = (int)ea[(size_t)e * 7 + 2];
      val = lr_emb[(size_t)lrid * LREW + j];
    }
    fij[a_swz(el, 6 + j)] = f32_bf16(val);
  }
  __syncthreads();

  // ---- h1 = relu(Fij @ W1^T + b1) : 16x64, K=64 ----
#pragma unroll
  for (int nt = 0; nt < EMHW / 16; ++nt) {
    v8f acc = {};
#pragma unroll
    for (int kt = 0; kt < FIJW / 32; ++kt) {
      v16bf a = *(const v16bf*)(fij + (kt * 32 + lane) * 16);
      v16bf b = *(const v16bf*)(B1 + ((nt * (FIJW / 32) + kt) * 32 + lane) * 16);
      acc = wmma_bf16(a, b, acc);
    }
    int m0 = (lane >> 4) * 8, n = nt * 16 + (lane & 15);
    float bias = b1[n];
    int half_s, i_s; kinv(n & 31, half_s, i_s);
    int sbase = (n >> 5) * 512 + half_s * 256 + i_s;
#pragma unroll
    for (int r = 0; r < 8; ++r)
      h1[sbase + (m0 + r) * 16] = f32_bf16(fmaxf(acc[r] + bias, 0.f));
  }
  __syncthreads();

  // ---- phi = relu(h1 @ W2^T + b2) : 16x32, K=64 ----
#pragma unroll
  for (int nt = 0; nt < EBDW / 16; ++nt) {
    v8f acc = {};
#pragma unroll
    for (int kt = 0; kt < EMHW / 32; ++kt) {
      v16bf a = *(const v16bf*)(h1 + (kt * 32 + lane) * 16);
      v16bf b = *(const v16bf*)(B2 + ((nt * (EMHW / 32) + kt) * 32 + lane) * 16);
      acc = wmma_bf16(a, b, acc);
    }
    int m0 = (lane >> 4) * 8, n = nt * 16 + (lane & 15);
    float bias = b2[n];
    int half_s, i_s; kinv(n & 31, half_s, i_s);
    int sbase = half_s * 256 + i_s;                 // n < 32 -> kt 0
#pragma unroll
    for (int r = 0; r < 8; ++r)
      phi[sbase + (m0 + r) * 16] = f32_bf16(fmaxf(acc[r] + bias, 0.f));
  }
  __syncthreads();

  // ---- bias_h = phi @ wb^T : 16x8 (N padded to 16), K=32 ----
  v8f bacc = {};
  {
    v16bf a = *(const v16bf*)(phi + lane * 16);
    v16bf b = *(const v16bf*)(Bwb + lane * 16);
    bacc = wmma_bf16(a, b, bacc);
  }

  // ---- dots straight from the D fragment: lane%16 = head, VGPR r = edge ----
  int h = lane & 15;
  if (h < HB) {
    int m0 = (lane >> 4) * 8;
    float scl = 1.0f / (4.0f * tau);   // sqrt(DK)=4
#pragma unroll
    for (int r = 0; r < 8; ++r) {
      int e = e0 + m0 + r;
      if (full || e < n_edges) {
        int s = edge_index[e];             // src
        int d = edge_index[n_edges + e];   // dst
        const float4* qp = (const float4*)(q    + (size_t)d * OUTD + h * DKH);
        const float4* kp = (const float4*)(kmat + (size_t)s * OUTD + h * DKH);
        float dot = 0.f;
#pragma unroll
        for (int t = 0; t < 4; ++t) {
          float4 a4 = qp[t], b4 = kp[t];
          dot += a4.x * b4.x + a4.y * b4.y + a4.z * b4.z + a4.w * b4.w;
        }
        float er = dot * scl + bacc[r];
        e_raw[(size_t)e * HB + h] = er;
        atomicAdd(&s1[d * HB + h], er);
        atomicAdd(&s2[d * HB + h], er * er);
        if (h == 0) atomicAdd(&cnt[d], 1.0f);
      }
    }
  }
}

// ---------------------------------------------------------------------------
// K3: per-(node,head) mean and 1/(std+eps).
// ---------------------------------------------------------------------------
__global__ void node_stats_kernel(const float* __restrict__ s1, const float* __restrict__ s2,
                                  const float* __restrict__ cnt, float* __restrict__ mean,
                                  float* __restrict__ inv, int n_nodes) {
  int i = blockIdx.x * blockDim.x + threadIdx.x;
  if (i >= n_nodes * HB) return;
  float c = fmaxf(cnt[i >> 3], 1.0f);
  float m = s1[i] / c;
  float var = fmaxf(s2[i] / c - m * m, 0.f);
  mean[i] = m;
  inv[i] = 1.0f / (sqrtf(var + 1e-6f) + 1e-6f);
}

// ---------------------------------------------------------------------------
// K4: e = tanh((e_raw - mean)/(std+eps)); ex = exp(e); denom += ex.
//     ex_out aliases e_in (in-place rewrite of the workspace buffer).
// ---------------------------------------------------------------------------
__global__ void edge_norm_kernel(const int* __restrict__ edge_index, const float* e_in,
                                 const float* __restrict__ mean, const float* __restrict__ inv,
                                 float* __restrict__ e_out, float* ex_out,
                                 float* __restrict__ denom, int n_edges) {
  int i = blockIdx.x * blockDim.x + threadIdx.x;
  if (i >= n_edges * HB) return;
  int e = i >> 3, h = i & 7;
  int d = edge_index[n_edges + e];
  float ev = tanhf((e_in[i] - mean[d * HB + h]) * inv[d * HB + h]);
  e_out[i] = ev;
  float xv = __expf(ev);        // e in [-1,1] -> max-free softmax is exact
  ex_out[i] = xv;
  atomicAdd(&denom[d * HB + h], xv);
}

// ---------------------------------------------------------------------------
// K5: out_accum[dst] += alpha * v[src]  (per edge, per head, 16 dims).
// ---------------------------------------------------------------------------
__global__ void edge_aggregate_kernel(const int* __restrict__ edge_index,
                                      const float* __restrict__ ex, const float* __restrict__ denom,
                                      const float* __restrict__ v, float* __restrict__ oacc,
                                      int n_edges) {
  int i = blockIdx.x * blockDim.x + threadIdx.x;
  if (i >= n_edges * HB) return;
  int e = i >> 3, h = i & 7;
  int s = edge_index[e], d = edge_index[n_edges + e];
  float alpha = ex[i] / (denom[d * HB + h] + 1e-12f);
  const float* vp = v + (size_t)s * OUTD + h * DKH;
  float* op = oacc + (size_t)d * OUTD + h * DKH;
#pragma unroll
  for (int t = 0; t < DKH; ++t) atomicAdd(&op[t], alpha * vp[t]);
}

// ---------------------------------------------------------------------------
// K6: out = out_accum @ Wp^T + bp  (WMMA, same shape as K1).
// ---------------------------------------------------------------------------
__global__ void __launch_bounds__(256)
proj_gemm_kernel(const float* __restrict__ acc_in, const __bf16* __restrict__ Bp,
                 const float* __restrict__ bp, float* __restrict__ out, int n_nodes) {
  __shared__ float xs[16 * OUTD];
  __shared__ __attribute__((aligned(32))) __bf16 as[16 * OUTD];
  int tile = blockIdx.x, tid = threadIdx.x;

  if (tile + 1 < gridDim.x)
    __builtin_prefetch(acc_in + (size_t)(tile + 1) * 16 * OUTD + tid * 8, 0, 1);

  stage_a_tile(acc_in, tile, n_nodes, xs, as, tid);

  int wave = tid >> 5, lane = tid & 31;
  v8f acc = {};
#pragma unroll
  for (int kt = 0; kt < OUTD / 32; ++kt) {
    v16bf a = *(const v16bf*)(as + (kt * 32 + lane) * 16);
    v16bf b = *(const v16bf*)(Bp + (((size_t)wave * (OUTD / 32) + kt) * 32 + lane) * 16);
    acc = wmma_bf16(a, b, acc);
  }
  int m0 = (lane >> 4) * 8, n = wave * 16 + (lane & 15);
  float bias = bp[n];
  if (tile * 16 + 16 <= n_nodes) {
    float* orow = out + (size_t)(tile * 16 + m0) * OUTD + n;
#pragma unroll
    for (int r = 0; r < 8; ++r) orow[(size_t)r * OUTD] = acc[r] + bias;
  } else {
#pragma unroll
    for (int r = 0; r < 8; ++r) {
      int row = tile * 16 + m0 + r;
      if (row < n_nodes) out[(size_t)row * OUTD + n] = acc[r] + bias;
    }
  }
}

// ---------------------------------------------------------------------------
extern "C" void kernel_launch(void* const* d_in, const int* in_sizes, int n_in,
                              void* d_out, int out_size, void* d_ws, size_t ws_size,
                              hipStream_t stream) {
  const float* x          = (const float*)d_in[0];
  const int*   edge_index = (const int*)  d_in[1];
  const float* ea         = (const float*)d_in[2];
  const float* Wq         = (const float*)d_in[3];
  const float* Wk         = (const float*)d_in[4];
  const float* Wv         = (const float*)d_in[5];
  const float* lr_emb     = (const float*)d_in[6];
  const float* W1         = (const float*)d_in[7];
  const float* b1         = (const float*)d_in[8];
  const float* W2         = (const float*)d_in[9];
  const float* b2         = (const float*)d_in[10];
  const float* wb         = (const float*)d_in[11];
  const float* Wp         = (const float*)d_in[12];
  const float* bp         = (const float*)d_in[13];
  const float* p_sigma    = (const float*)d_in[14];
  const int*   p_log1p    = (const int*)  d_in[15];
  const float* p_cscale   = (const float*)d_in[16];
  const float* p_tau      = (const float*)d_in[17];

  int N = in_sizes[0] / IND;   // 50000
  int E = in_sizes[1] / 2;     // 800000

  // Workspace carve (floats). ~137 MB for reference sizes.
  float* w = (float*)d_ws;
  size_t nf = (size_t)N * OUTD;
  float* q     = w; w += nf;
  float* kk    = w; w += nf;
  float* vv    = w; w += nf;
  float* oacc  = w; w += nf;
  float* e_raw = w; w += (size_t)E * HB;   // reused as ex after K4
  float* s1    = w; w += (size_t)N * HB;
  float* s2    = w; w += (size_t)N * HB;
  float* cnt   = w; w += (size_t)N;
  float* mean  = w; w += (size_t)N * HB;
  float* inv   = w; w += (size_t)N * HB;
  float* denom = w; w += (size_t)N * HB;
  // packed bf16 weights (fragment-ordered), sizes in floats (2 bf16 per float)
  __bf16* Bq  = (__bf16*)w; w += 8192;     // 8nt*4kt*512
  __bf16* Bk  = (__bf16*)w; w += 8192;
  __bf16* Bv  = (__bf16*)w; w += 8192;
  __bf16* Bp  = (__bf16*)w; w += 8192;
  __bf16* B1  = (__bf16*)w; w += 2048;     // 4nt*2kt*512
  __bf16* B2  = (__bf16*)w; w += 1024;     // 2nt*2kt*512
  __bf16* Bwb = (__bf16*)w; w += 256;      // 1nt*1kt*512

  float* out_x = (float*)d_out;   // N x 128
  float* out_e = out_x + nf;      // E x 8

  // Zero atomic accumulators (capture-safe stream memsets).
  hipMemsetAsync(oacc, 0, nf * sizeof(float), stream);
  hipMemsetAsync(s1, 0, ((size_t)N * HB * 2 + (size_t)N) * sizeof(float), stream);
  hipMemsetAsync(denom, 0, (size_t)N * HB * sizeof(float), stream);

  // Weight packing (tiny; off the critical path).
  pack_b_kernel<<<64, 256, 0, stream>>>(Wq, IND, IND, OUTD, 4, 8, Bq);
  pack_b_kernel<<<64, 256, 0, stream>>>(Wk, IND, IND, OUTD, 4, 8, Bk);
  pack_b_kernel<<<64, 256, 0, stream>>>(Wv, IND, IND, OUTD, 4, 8, Bv);
  pack_b_kernel<<<64, 256, 0, stream>>>(Wp, OUTD, OUTD, OUTD, 4, 8, Bp);
  pack_b_kernel<<<16, 256, 0, stream>>>(W1, FIJV, FIJV, EMHW, 2, 4, B1);
  pack_b_kernel<<< 8, 256, 0, stream>>>(W2, EMHW, EMHW, EBDW, 2, 2, B2);
  pack_b_kernel<<< 2, 256, 0, stream>>>(wb, EBDW, EBDW, HB, 1, 1, Bwb);

  int ntiles = (N + 15) / 16;
  qkv_gemm_kernel<<<dim3(ntiles, 3), 256, 0, stream>>>(x, Bq, Bk, Bv, q, kk, vv, N);

  int eblocks = (E + 127) / 128;
  edge_mlp_dot_kernel<<<eblocks, 256, 0, stream>>>(edge_index, ea, lr_emb, B1, b1, B2, b2, Bwb,
                                                   q, kk, p_sigma, p_log1p, p_cscale, p_tau,
                                                   e_raw, s1, s2, cnt, E);

  node_stats_kernel<<<(N * HB + 255) / 256, 256, 0, stream>>>(s1, s2, cnt, mean, inv, N);

  edge_norm_kernel<<<(int)(((size_t)E * HB + 255) / 256), 256, 0, stream>>>(
      edge_index, e_raw, mean, inv, out_e, e_raw, denom, E);

  edge_aggregate_kernel<<<(int)(((size_t)E * HB + 255) / 256), 256, 0, stream>>>(
      edge_index, e_raw, denom, vv, oacc, E);

  proj_gemm_kernel<<<ntiles, 256, 0, stream>>>(oacc, Bp, bp, out_x, N);
}
